// GNN_8177617732272
// MI455X (gfx1250) — matrix operand order repaired and checked by reading, not compiled, not run on hardware
//
#include <hip/hip_runtime.h>

// ---------------------------------------------------------------------------
// GNN (2-layer GIN) for MI455X / gfx1250.
// Dominant cost: adj[16384x16384] fp32 streamed from HBM twice (~92us floor
// at 23.3 TB/s). Aggregation uses v_wmma_f32_16x16x32_bf16 with an exact
// bf16 adj (0/1 values) and a hi/lo bf16 split of h for ~fp32 accuracy.
// deg = adj @ ones is fused as an extra WMMA against a constant ones tile.
// v3: split-K=4 (4096 waves), double-buffered async LDS staging of B tiles,
// A tile software-pipelined two k-steps ahead (~4KB outstanding per wave).
// ---------------------------------------------------------------------------

typedef __attribute__((ext_vector_type(16))) __bf16 v16bf;
typedef __attribute__((ext_vector_type(8)))  __bf16 v8bf;
typedef __attribute__((ext_vector_type(8)))  float  v8f;
typedef int v4i_ __attribute__((vector_size(16)));   // matches async builtin param

#define NNODES 16384
#define HID 64
#define NSPLIT 4        // split-K chunks
#define KCHUNK (NNODES / NSPLIT)
#define KSTAGE 64       // k-elements staged to LDS per barrier phase
#define LSTRIDE 72      // LDS elements per column (36-dword stride, no bank conflicts)

#if defined(__HIP_DEVICE_COMPILE__) && __has_builtin(__builtin_amdgcn_global_load_async_to_lds_b128)
#define USE_ASYNC_LDS 1
#else
#define USE_ASYNC_LDS 0
#endif

__device__ __forceinline__ v8f wmma_bf16(v16bf a, v16bf b, v8f c) {
  // (neg_a, A, neg_b, B, c_mod, C, reuse_a, reuse_b)
  return __builtin_amdgcn_wmma_f32_16x16x32_bf16(false, a, false, b, (short)0, c,
                                                 false, false);
}

#if USE_ASYNC_LDS
__device__ __forceinline__ __attribute__((address_space(1))) v4i_* as_gv4(
    const void* p) {
  return (__attribute__((address_space(1))) v4i_*)(unsigned long long)(uintptr_t)p;
}
__device__ __forceinline__ __attribute__((address_space(3))) v4i_* as_lv4(
    const void* p) {
  // generic LDS pointer: low 32 bits are the LDS byte address
  return (__attribute__((address_space(3))) v4i_*)(unsigned int)(uintptr_t)p;
}
#endif

// copy 32 bytes global -> LDS (per staging thread)
__device__ __forceinline__ void stage32(const __bf16* g, __bf16* l) {
#if USE_ASYNC_LDS
  __builtin_amdgcn_global_load_async_to_lds_b128(as_gv4(g), as_lv4(l), 0, 0);
  __builtin_amdgcn_global_load_async_to_lds_b128(as_gv4(g), as_lv4(l), 16, 0);
#else
  uint4 v0 = ((const uint4*)g)[0];
  uint4 v1 = ((const uint4*)g)[1];
  ((uint4*)l)[0] = v0;
  ((uint4*)l)[1] = v1;
#endif
}

__device__ __forceinline__ void stage_wait() {
#if USE_ASYNC_LDS
  asm volatile("s_wait_asynccnt 0" ::: "memory");
#endif
}

__device__ __forceinline__ v16bf ldsB_read(const __bf16* p) {
  v8bf lo = *(const v8bf*)p;        // 16B-aligned ds_load
  v8bf hi = *(const v8bf*)(p + 8);
  return __builtin_shufflevector(lo, hi, 0, 1, 2, 3, 4, 5, 6, 7, 8, 9, 10, 11,
                                 12, 13, 14, 15);
}

struct ATile { float4 f0, f1, f2, f3; };

__device__ __forceinline__ ATile load_a(const float* arow, int k, int o0, int o1) {
  ATile t;
  t.f0 = *(const float4*)(arow + k + o0);
  t.f1 = *(const float4*)(arow + k + o0 + 4);
  t.f2 = *(const float4*)(arow + k + o1);
  t.f3 = *(const float4*)(arow + k + o1 + 4);
  return t;
}

__device__ __forceinline__ v16bf pack_a(const ATile& t) {
  v16bf a;
  a[0]  = (__bf16)t.f0.x; a[1]  = (__bf16)t.f0.y; a[2]  = (__bf16)t.f0.z; a[3]  = (__bf16)t.f0.w;
  a[4]  = (__bf16)t.f1.x; a[5]  = (__bf16)t.f1.y; a[6]  = (__bf16)t.f1.z; a[7]  = (__bf16)t.f1.w;
  a[8]  = (__bf16)t.f2.x; a[9]  = (__bf16)t.f2.y; a[10] = (__bf16)t.f2.z; a[11] = (__bf16)t.f2.w;
  a[12] = (__bf16)t.f3.x; a[13] = (__bf16)t.f3.y; a[14] = (__bf16)t.f3.z; a[15] = (__bf16)t.f3.w;
  return a;
}

// ---------------------------------------------------------------------------
// K0: transpose + hi/lo bf16 split:  src[N][64] f32 -> bth/btl [64][N] bf16
// ---------------------------------------------------------------------------
__global__ __launch_bounds__(256) void split_transpose(
    const float* __restrict__ src, __bf16* __restrict__ bth,
    __bf16* __restrict__ btl) {
  __shared__ __bf16 shh[64][65];
  __shared__ __bf16 shl[64][65];
  const int tx = threadIdx.x;        // 0..63
  const int ty = threadIdx.y;        // 0..3
  const int iBase = blockIdx.x * 64; // 256 blocks cover 16384 rows
  for (int rr = ty; rr < 64; rr += 4) {
    float v = src[(size_t)(iBase + rr) * HID + tx];
    __bf16 hi = (__bf16)v;
    float lo = v - (float)hi;
    shh[tx][rr] = hi;
    shl[tx][rr] = (__bf16)lo;
  }
  __syncthreads();
  for (int cc = ty; cc < 64; cc += 4) {
    bth[(size_t)cc * NNODES + iBase + tx] = shh[cc][tx];
    btl[(size_t)cc * NNODES + iBase + tx] = shl[cc][tx];
  }
}

// ---------------------------------------------------------------------------
// K1: rawPart[c][N][64] = adj[:, chunk c] @ (hi+lo); degPart[c][N] = row sums.
// Grid: 512 blocks x 256 threads. Block = (row group of 128) x (K chunk of 4096).
// 8 waves/block share double-buffered LDS B tiles; one wave owns 16 rows.
// ---------------------------------------------------------------------------
__global__ __launch_bounds__(256) void gemm_adj(
    const float* __restrict__ adj,   // [N][N] fp32 (0/1 values)
    const __bf16* __restrict__ bth,  // [64][N] hi part of h^T
    const __bf16* __restrict__ btl,  // [64][N] lo part of h^T
    float* __restrict__ rawPart,     // [NSPLIT][N][64]
    float* __restrict__ degPart)     // [NSPLIT][N]
{
  __shared__ __bf16 ldsB[2][2][64 * LSTRIDE];  // [slot][hi/lo][col*LSTRIDE+k]

  const int tid = threadIdx.x;
  const int chunk = blockIdx.x & (NSPLIT - 1);
  const int sg = blockIdx.x / NSPLIT;  // 0..127 row group
  const int kStart = chunk * KCHUNK;
  const int kEnd = kStart + KCHUNK;

  const int wave = tid >> 5;
  const int lane = tid & 31;
  const int half = lane >> 4;
  const int lr = lane & 15;
  const int rowBase = sg * 128 + wave * 16;

  // staging map: thread -> (column, 16-element quarter)
  const int scol = tid >> 2;
  const int spart = tid & 3;
  const size_t gOff = (size_t)scol * NNODES + spart * 16;
  const int lOff = scol * LSTRIDE + spart * 16;

  const float* arow = adj + (size_t)(rowBase + lr) * NNODES;
  const int o0 = half * 16;      // A: K = k+o0 .. k+o0+7
  const int o1 = half * 16 + 8;  // A: K = k+o1 .. k+o1+7

  v8f z = {};
  v8f acc[4] = {z, z, z, z};
  v8f accd = z;  // degree accumulator (every column == row degree)

  v16bf ones;
#pragma unroll
  for (int e = 0; e < 16; ++e) ones[e] = (__bf16)1.0f;

  // prologue: stage first slab, prefetch first two A tiles (depth-2 pipeline)
  stage32(bth + gOff + kStart, &ldsB[0][0][lOff]);
  stage32(btl + gOff + kStart, &ldsB[0][1][lOff]);
  ATile aCur = load_a(arow, kStart, o0, o1);
  ATile aNxt = load_a(arow, kStart + 32, o0, o1);
  stage_wait();
  __syncthreads();

  int slot = 0;
  for (int kb = kStart; kb < kEnd; kb += KSTAGE) {
    const int kbn = kb + KSTAGE;
    if (kbn < kEnd) {  // stage next slab into the other buffer (async)
      stage32(bth + gOff + kbn, &ldsB[slot ^ 1][0][lOff]);
      stage32(btl + gOff + kbn, &ldsB[slot ^ 1][1][lOff]);
    }
#pragma unroll
    for (int s = 0; s < 2; ++s) {  // two 32-wide k-steps per slab
      const int k = kb + s * 32;
      int kp = k + 64;                 // prefetch two steps ahead
      if (kp >= kEnd) kp = kStart;     // dummy (in-bounds, value unused)
      ATile aInc = load_a(arow, kp, o0, o1);

      v16bf a = pack_a(aCur);
      const int klocal = s * 32 + half * 16;
#pragma unroll
      for (int n = 0; n < 4; ++n) {
        v16bf tbh = ldsB_read(&ldsB[slot][0][(n * 16 + lr) * LSTRIDE + klocal]);
        acc[n] = wmma_bf16(a, tbh, acc[n]);
        v16bf tbl = ldsB_read(&ldsB[slot][1][(n * 16 + lr) * LSTRIDE + klocal]);
        acc[n] = wmma_bf16(a, tbl, acc[n]);
      }
      accd = wmma_bf16(a, ones, accd);

      aCur = aNxt;
      aNxt = aInc;
    }
    stage_wait();     // our async stores to the other slot are done
    __syncthreads();  // everyone finished reading current slot
    slot ^= 1;
  }

  // Write-out: C layout -> VGPR r holds row (half*8 + r), lane holds column lr.
  const int orow = rowBase + half * 8;
  float* raw = rawPart + (size_t)chunk * NNODES * HID;
#pragma unroll
  for (int r = 0; r < 8; ++r) {
    size_t base = (size_t)(orow + r) * HID + lr;
    raw[base + 0]  = acc[0][r];
    raw[base + 16] = acc[1][r];
    raw[base + 32] = acc[2][r];
    raw[base + 48] = acc[3][r];
  }
  if (lr == 0) {
#pragma unroll
    for (int r = 0; r < 8; ++r) degPart[chunk * NNODES + orow + r] = accd[r];
  }
}

// ---------------------------------------------------------------------------
// K1b: pooled[i][j] = sum_c raw[c][i][j] / sum_c deg[c][i]
// ---------------------------------------------------------------------------
__global__ __launch_bounds__(256) void reduce_pooled(
    const float* __restrict__ rawPart, const float* __restrict__ degPart,
    float* __restrict__ pooled) {
  const int idx = blockIdx.x * 256 + threadIdx.x;  // 4096 blocks -> N*64
  const int row = idx >> 6;
  float d = 0.0f, v = 0.0f;
#pragma unroll
  for (int c = 0; c < NSPLIT; ++c) {
    d += degPart[c * NNODES + row];
    v += rawPart[(size_t)c * NNODES * HID + idx];
  }
  pooled[idx] = v / d;
}

// ---------------------------------------------------------------------------
// K2/K4: out[N][64] = f(in) @ W + bias, where f = identity or BN+ReLU.
// ---------------------------------------------------------------------------
__global__ __launch_bounds__(256) void mlp_gemm(
    const float* __restrict__ in, const float* __restrict__ W,
    const float* __restrict__ bias, const float* __restrict__ mean,
    const float* __restrict__ rstd, const float* __restrict__ gamma,
    const float* __restrict__ beta, int doBN, float* __restrict__ out) {
  __shared__ float shW[64][64];
  __shared__ float shP[16][64];
  const int tx = threadIdx.x;  // column j
  const int ty = threadIdx.y;  // 0..3
  const int tid = ty * 64 + tx;
  for (int idx = tid; idx < 64 * 64; idx += 256) shW[idx >> 6][idx & 63] = W[idx];
  const int rowBase = blockIdx.x * 16;
  for (int rr = ty; rr < 16; rr += 4) {
    float v = in[(size_t)(rowBase + rr) * HID + tx];
    if (doBN) {
      v = gamma[tx] * (v - mean[tx]) * rstd[tx] + beta[tx];
      v = v > 0.0f ? v : 0.0f;
    }
    shP[rr][tx] = v;
  }
  __syncthreads();
  float acc[4];
#pragma unroll
  for (int r = 0; r < 4; ++r) acc[r] = bias[tx];
  for (int k = 0; k < 64; ++k) {
    float w = shW[k][tx];
#pragma unroll
    for (int r = 0; r < 4; ++r) acc[r] += shP[ty * 4 + r][k] * w;
  }
#pragma unroll
  for (int r = 0; r < 4; ++r)
    out[(size_t)(rowBase + ty * 4 + r) * HID + tx] = acc[r];
}

// ---------------------------------------------------------------------------
// K3/K5: per-column mean and rsqrt(var+eps) over 16384 rows (deterministic).
// ---------------------------------------------------------------------------
__global__ __launch_bounds__(256) void col_stats(const float* __restrict__ t,
                                                 float* __restrict__ mean,
                                                 float* __restrict__ rstd) {
  const int col = blockIdx.x;
  const int tid = threadIdx.x;
  float s = 0.0f, s2 = 0.0f;
  for (int i = tid; i < NNODES; i += 256) {
    float v = t[(size_t)i * HID + col];
    s += v;
    s2 += v * v;
  }
  __shared__ float sh[256];
  __shared__ float sh2[256];
  sh[tid] = s;
  sh2[tid] = s2;
  __syncthreads();
  for (int off = 128; off > 0; off >>= 1) {
    if (tid < off) {
      sh[tid] += sh[tid + off];
      sh2[tid] += sh2[tid + off];
    }
    __syncthreads();
  }
  if (tid == 0) {
    float m = sh[0] * (1.0f / NNODES);
    float var = sh2[0] * (1.0f / NNODES) - m * m;
    mean[col] = m;
    rstd[col] = rsqrtf(var + 1e-5f);
  }
}

// ---------------------------------------------------------------------------
// K6: h = relu(gamma*(t2-mean)*rstd + beta), written to two destinations.
// ---------------------------------------------------------------------------
__global__ __launch_bounds__(256) void bn_relu(
    const float* __restrict__ t2, const float* __restrict__ mean,
    const float* __restrict__ rstd, const float* __restrict__ gamma,
    const float* __restrict__ beta, float* __restrict__ out1,
    float* __restrict__ out2) {
  const int idx = blockIdx.x * 256 + threadIdx.x;  // 4096 blocks -> N*64
  const int col = idx & 63;
  float v = t2[idx];
  v = gamma[col] * (v - mean[col]) * rstd[col] + beta[col];
  v = v > 0.0f ? v : 0.0f;
  out1[idx] = v;
  out2[idx] = v;
}

// ---------------------------------------------------------------------------
// K7: graph readout (graph_pool row g is nonzero only on its 128 nodes).
// ---------------------------------------------------------------------------
__global__ __launch_bounds__(64) void graph_readout(
    const float* __restrict__ gp,  // [128][16384]
    const float* __restrict__ h,   // [16384][64]
    float* __restrict__ out) {     // [128][64]
  const int g = blockIdx.x;
  const int j = threadIdx.x;
  const int nb = g * 128;
  float acc = 0.0f;
  for (int i = 0; i < 128; ++i)
    acc += gp[(size_t)g * NNODES + nb + i] * h[(size_t)(nb + i) * HID + j];
  out[g * HID + j] = acc;
}

// ---------------------------------------------------------------------------
extern "C" void kernel_launch(void* const* d_in, const int* in_sizes, int n_in,
                              void* d_out, int out_size, void* d_ws,
                              size_t ws_size, hipStream_t stream) {
  (void)in_sizes; (void)n_in; (void)out_size; (void)ws_size;

  const float* x   = (const float*)d_in[0];
  const float* adj = (const float*)d_in[1];
  const float* gp  = (const float*)d_in[2];
  const float* W1[2]  = {(const float*)d_in[3],  (const float*)d_in[11]};
  const float* b1[2]  = {(const float*)d_in[4],  (const float*)d_in[12]};
  const float* g1[2]  = {(const float*)d_in[5],  (const float*)d_in[13]};
  const float* be1[2] = {(const float*)d_in[6],  (const float*)d_in[14]};
  const float* W2[2]  = {(const float*)d_in[7],  (const float*)d_in[15]};
  const float* b2[2]  = {(const float*)d_in[8],  (const float*)d_in[16]};
  const float* go[2]  = {(const float*)d_in[9],  (const float*)d_in[17]};
  const float* beo[2] = {(const float*)d_in[10], (const float*)d_in[18]};

  char* ws = (char*)d_ws;
  __bf16* bth   = (__bf16*)(ws);                      // 2 MB    h^T hi
  __bf16* btl   = (__bf16*)(ws + (size_t)(2 << 20));  // 2 MB    h^T lo
  float* raw    = (float*)(ws + (size_t)(4 << 20));   // 16 MB   [4][N][64]
  float* degp   = (float*)(ws + (size_t)(20 << 20));  // 256 KB  [4][N]
  float* pooled = (float*)(ws + (size_t)(21 << 20));  // 4 MB
  float* t      = (float*)(ws + (size_t)(25 << 20));  // 4 MB
  float* t2     = (float*)(ws + (size_t)(29 << 20));  // 4 MB
  float* h      = (float*)(ws + (size_t)(33 << 20));  // 4 MB
  float* m1     = (float*)(ws + (size_t)(37 << 20));  // stats: 4x64 floats
  float* rs1 = m1 + 64;
  float* m2  = m1 + 128;
  float* rs2 = m1 + 192;

  float* out_pool  = (float*)d_out;        // [128][64]
  float* out_nodes = out_pool + 128 * 64;  // [16384][64]

  const float* cur = x;
  for (int l = 0; l < 2; ++l) {
    split_transpose<<<256, dim3(64, 4), 0, stream>>>(cur, bth, btl);
    gemm_adj<<<128 * NSPLIT, 256, 0, stream>>>(adj, bth, btl, raw, degp);
    reduce_pooled<<<4096, 256, 0, stream>>>(raw, degp, pooled);
    mlp_gemm<<<1024, dim3(64, 4), 0, stream>>>(pooled, W1[l], b1[l], nullptr,
                                               nullptr, nullptr, nullptr, 0, t);
    col_stats<<<64, 256, 0, stream>>>(t, m1, rs1);
    mlp_gemm<<<1024, dim3(64, 4), 0, stream>>>(t, W2[l], b2[l], m1, rs1, g1[l],
                                               be1[l], 1, t2);
    col_stats<<<64, 256, 0, stream>>>(t2, m2, rs2);
    float* o2 = (l == 0) ? h : out_nodes;
    bn_relu<<<4096, 256, 0, stream>>>(t2, m2, rs2, go[l], beo[l], h, o2);
    cur = h;
  }
  graph_readout<<<128, 64, 0, stream>>>(gp, h, out_pool);
}
